// PoseNet_88295937671374
// MI455X (gfx1250) — matrix operand order repaired and checked by reading, not compiled, not run on hardware
//
#include <hip/hip_runtime.h>
#include <hip/hip_bf16.h>
#include <math.h>

// ---------------------------------------------------------------------------
// Types for CDNA5 WMMA
// ---------------------------------------------------------------------------
typedef __attribute__((ext_vector_type(16))) __bf16 v16bf;
typedef __attribute__((ext_vector_type(8)))  __bf16 v8bf;
typedef __attribute__((ext_vector_type(4)))  __bf16 v4bf;
typedef __attribute__((ext_vector_type(8)))  float  v8f;
typedef int vi4 __attribute__((vector_size(16)));   // int4 vector for async-LDS builtin

#define BM 128
#define BN 64
#define BK 32
#define LDSTRIDE 40          // row stride (elements): 80 bytes -> 16B-aligned chunks
#define CONV_THREADS 256

#define NANCH 49104
#define KTOP  1000
#define KP_SZ (18 * 128 * 128)
#define SCORE_THR 0.05f
#define IOU_THR   0.5f

#if __has_builtin(__builtin_amdgcn_global_load_async_to_lds_b128)
#define ASYNC_B_TILE 1
#endif

// ---------------------------------------------------------------------------
// f32 -> bf16 conversion (float4 in, 8B bf16 out); n must be a multiple of 4.
// ---------------------------------------------------------------------------
__global__ void k_cvt_bf16(const float* __restrict__ in, __bf16* __restrict__ out, int n4)
{
  int i = blockIdx.x * blockDim.x + threadIdx.x;
  if (i >= n4) return;
  float4 v = ((const float4*)in)[i];
  v4bf o;
  o[0] = (__bf16)v.x; o[1] = (__bf16)v.y; o[2] = (__bf16)v.z; o[3] = (__bf16)v.w;
  ((v4bf*)out)[i] = o;
}

// ---------------------------------------------------------------------------
// Implicit-GEMM conv (NCHW, OIHW weights), bf16 in / bf16 out, f32 accumulate.
//   M = D*D pixels, N = OC, K = C*KH*KW (multiple of 32 for every layer here)
// Templated on KH,KW and log2(D) so all addressing is shifts/constant-divides.
// Block: 256 threads = 8 waves; block tile 128(M) x 64(N).
// Wave (wm 0..3, wn 0..1) computes a 32x32 C tile: 4 v_wmma per K-step.
// B tile staged via async global->LDS (ASYNCcnt) when available, overlapping
// with the VALU-heavy im2col A staging.
// ---------------------------------------------------------------------------
template <int KH, int KW, int LOG2D>
__global__ __launch_bounds__(CONV_THREADS)
void k_conv_wmma_t(const __bf16* __restrict__ X, const __bf16* __restrict__ Wt,
                   const float* __restrict__ Bias, __bf16* __restrict__ Y,
                   int C, int OC, int relu)
{
  constexpr int D    = 1 << LOG2D;
  constexpr int P    = D * D;
  constexpr int KHW  = KH * KW;
  constexpr int PAD  = (KH == 3) ? 1 : 0;
  const int KTOT = C * KHW;

  __shared__ __bf16 As[BM][LDSTRIDE];   // [pixel][k]
  __shared__ __bf16 Bs[BN][LDSTRIDE];   // [oc][k]  (N-major for contiguous frags)

  const int tid    = threadIdx.x;
  const int lane   = tid & 31;
  const int wave   = tid >> 5;
  const int wm     = wave & 3;          // M wave offset: wm*32
  const int wn     = wave >> 2;         // N wave offset: wn*32
  const int pTile  = blockIdx.x * BM;
  const int ocTile = blockIdx.y * BN;
  const int hi     = lane >> 4;         // lane half (0/1)
  const int l16    = lane & 15;

  // A staging: this thread owns column kk for rows rowBase + j*8
  const int kkA     = tid & 31;
  const int rowBase = tid >> 5;
  // B staging: this thread owns 8 contiguous k at (n, kchunk)
  const int nB      = tid >> 2;
  const int kchunk  = (tid & 3) * 8;
  const int ocB     = ocTile + nB;

  v8f acc00 = {}, acc01 = {}, acc10 = {}, acc11 = {};

  for (int k0 = 0; k0 < KTOT; k0 += BK) {
    // ---- stage B (bf16 weights): async global->LDS, 16B per thread -------
#ifdef ASYNC_B_TILE
    if (ocB < OC) {
      __builtin_amdgcn_global_load_async_to_lds_b128(
          (__attribute__((address_space(1))) vi4*)(Wt + (size_t)ocB * KTOT + k0 + kchunk),
          (__attribute__((address_space(3))) vi4*)&Bs[nB][kchunk], 0, 0);
    } else {
      *(v8bf*)&Bs[nB][kchunk] = (v8bf)(__bf16)0.0f;
    }
#else
    {
      v8bf wb = (ocB < OC) ? *(const v8bf*)(Wt + (size_t)ocB * KTOT + k0 + kchunk)
                           : (v8bf)(__bf16)0.0f;
      *(v8bf*)&Bs[nB][kchunk] = wb;
    }
#endif
    // ---- stage A (im2col, bf16): per-thread constant (ic,kh,kw) ----------
    {
      const int k  = k0 + kkA;
      const int ic = k / KHW;              // constant divisor -> mul/shift
      const int r  = k - ic * KHW;
      const int kh = r / KW;
      const int kw = r - kh * KW;
      const __bf16* __restrict__ Xc = X + (size_t)ic * P;
#pragma unroll
      for (int j = 0; j < BM / 8; ++j) {
        const int row = rowBase + j * 8;
        const int p   = pTile + row;
        const int py  = p >> LOG2D;
        const int px  = p & (D - 1);
        const int iy  = py + kh - PAD;
        const int ix  = px + kw - PAD;
        __bf16 v = (__bf16)0.0f;
        if (p < P && iy >= 0 && iy < D && ix >= 0 && ix < D)
          v = Xc[(iy << LOG2D) + ix];
        As[row][kkA] = v;
      }
    }
    // prefetch next weight K-slab (global_prefetch_b8; speculative)
    if (k0 + BK < KTOT)
      __builtin_prefetch(&Wt[(size_t)(ocTile + lane) * KTOT + k0 + BK], 0, 1);
#ifdef ASYNC_B_TILE
#if __has_builtin(__builtin_amdgcn_s_wait_asynccnt)
    __builtin_amdgcn_s_wait_asynccnt(0);
#else
    asm volatile("s_wait_asynccnt 0" ::: "memory");
#endif
#endif
    __syncthreads();

    // ---- fragments: contiguous 16B LDS chunks (ds_load_b128) -------------
    // A (ISA 7.12.2): lane -> M = l16; K runs [hi*8,+8) and [16+hi*8,+8)
    const int mrow0 = wm * 32 + l16;
    const int mrow1 = mrow0 + 16;
    v8bf a0lo = *(const v8bf*)&As[mrow0][hi * 8];
    v8bf a0hi = *(const v8bf*)&As[mrow0][16 + hi * 8];
    v8bf a1lo = *(const v8bf*)&As[mrow1][hi * 8];
    v8bf a1hi = *(const v8bf*)&As[mrow1][16 + hi * 8];
    v16bf a0 = __builtin_shufflevector(a0lo, a0hi, 0,1,2,3,4,5,6,7,8,9,10,11,12,13,14,15);
    v16bf a1 = __builtin_shufflevector(a1lo, a1hi, 0,1,2,3,4,5,6,7,8,9,10,11,12,13,14,15);
    // B: lane -> N = l16; K runs [hi*8,+8) and [16+hi*8,+8) in Bs[n][*]
    const int nc0 = wn * 32 + l16;
    const int nc1 = nc0 + 16;
    v8bf b0lo = *(const v8bf*)&Bs[nc0][hi * 8];
    v8bf b0hi = *(const v8bf*)&Bs[nc0][16 + hi * 8];
    v8bf b1lo = *(const v8bf*)&Bs[nc1][hi * 8];
    v8bf b1hi = *(const v8bf*)&Bs[nc1][16 + hi * 8];
    v16bf b0 = __builtin_shufflevector(b0lo, b0hi, 0,1,2,3,4,5,6,7,8,9,10,11,12,13,14,15);
    v16bf b1 = __builtin_shufflevector(b1lo, b1hi, 0,1,2,3,4,5,6,7,8,9,10,11,12,13,14,15);

    acc00 = __builtin_amdgcn_wmma_f32_16x16x32_bf16(false, a0, false, b0, (short)0, acc00, false, false);
    acc01 = __builtin_amdgcn_wmma_f32_16x16x32_bf16(false, a0, false, b1, (short)0, acc01, false, false);
    acc10 = __builtin_amdgcn_wmma_f32_16x16x32_bf16(false, a1, false, b0, (short)0, acc10, false, false);
    acc11 = __builtin_amdgcn_wmma_f32_16x16x32_bf16(false, a1, false, b1, (short)0, acc11, false, false);
    __syncthreads();
  }

  // ---- epilogue: lane holds 8 consecutive pixels per tile -> one v8bf ----
  // C/D layout: element r -> M = r + hi*8, N = l16
  const int pb0 = pTile + wm * 32 + hi * 8;        // M base, frags +0 / +16
  const int no0 = ocTile + wn * 32 + l16;          // N, frags +0 / +16
#pragma unroll
  for (int mi = 0; mi < 2; ++mi) {
    const int p0 = pb0 + mi * 16;
    if (p0 >= P) continue;                          // P is a multiple of 16
    v8f am0 = (mi == 0) ? acc00 : acc10;
    v8f am1 = (mi == 0) ? acc01 : acc11;
#pragma unroll
    for (int ni = 0; ni < 2; ++ni) {
      const int n = no0 + ni * 16;
      if (n >= OC) continue;
      v8f av = (ni == 0) ? am0 : am1;
      const float bsv = Bias[n];
      v8bf ov;
#pragma unroll
      for (int r = 0; r < 8; ++r) {
        float v = av[r] + bsv;
        if (relu) v = fmaxf(v, 0.0f);
        ov[r] = (__bf16)v;
      }
      *(v8bf*)(Y + (size_t)n * P + p0) = ov;
    }
  }
}

// ---------------------------------------------------------------------------
// Nearest-upsample + channel concat for the keypoint branch (bf16).
// ---------------------------------------------------------------------------
__global__ void k_upcat(const __bf16* __restrict__ k5, const __bf16* __restrict__ k4,
                        const __bf16* __restrict__ k3, const __bf16* __restrict__ k2,
                        __bf16* __restrict__ cat)
{
  int idx = blockIdx.x * blockDim.x + threadIdx.x;
  const int total = 512 * 128 * 128;
  if (idx >= total) return;
  int c = idx >> 14;
  int p = idx & 16383;
  int y = p >> 7, x = p & 127;
  __bf16 v;
  if (c < 128)       v = k5[c * 256 + (y >> 3) * 16 + (x >> 3)];
  else if (c < 256)  v = k4[(c - 128) * 1024 + (y >> 2) * 32 + (x >> 2)];
  else if (c < 384)  v = k3[(c - 256) * 4096 + (y >> 1) * 64 + (x >> 1)];
  else               v = k2[(c - 384) * 16384 + y * 128 + x];
  cat[idx] = v;
}

// ---------------------------------------------------------------------------
// Anchor generation + box decode + clip + sigmoid scores + threshold mask.
// Reads bf16 head outputs, produces f32 boxes/scores.
// ---------------------------------------------------------------------------
__global__ void k_decode(const __bf16* __restrict__ r3, const __bf16* __restrict__ r4,
                         const __bf16* __restrict__ r5, const __bf16* __restrict__ r6,
                         const __bf16* __restrict__ r7,
                         const __bf16* __restrict__ c3, const __bf16* __restrict__ c4,
                         const __bf16* __restrict__ c5, const __bf16* __restrict__ c6,
                         const __bf16* __restrict__ c7,
                         float* __restrict__ boxes, float* __restrict__ scores,
                         float* __restrict__ masked)
{
  int n = blockIdx.x * blockDim.x + threadIdx.x;
  if (n >= NANCH) return;

  int lvl, off;
  if      (n < 36864) { lvl = 0; off = 0;     }
  else if (n < 46080) { lvl = 1; off = 36864; }
  else if (n < 48384) { lvl = 2; off = 46080; }
  else if (n < 48960) { lvl = 3; off = 48384; }
  else                { lvl = 4; off = 48960; }

  const int   fwt[5]    = {64, 32, 16, 8, 4};
  const float strid[5]  = {8.f, 16.f, 32.f, 64.f, 128.f};
  const float basesz[5] = {32.f, 64.f, 128.f, 256.f, 512.f};

  const __bf16* rp; const __bf16* cp;
  if      (lvl == 0) { rp = r3; cp = c3; }
  else if (lvl == 1) { rp = r4; cp = c4; }
  else if (lvl == 2) { rp = r5; cp = c5; }
  else if (lvl == 3) { rp = r6; cp = c6; }
  else               { rp = r7; cp = c7; }

  int fw   = fwt[lvl];
  int hwA  = fw * fw;
  int loc  = n - off;
  int s    = loc / 9;
  int a    = loc - s * 9;
  int gy   = s / fw;
  int gx   = s - gy * fw;

  float stride = strid[lvl];
  float sx = (gx + 0.5f) * stride;
  float sy = (gy + 0.5f) * stride;

  const float rats[3] = {0.5f, 1.0f, 2.0f};
  const float scls[3] = {1.0f, 1.2599210499f, 1.5874010520f};
  int ai = a / 3;
  float ratio = rats[ai];
  float scale = scls[a - ai * 3];
  float wh    = basesz[lvl] * scale;
  float area  = wh * wh;
  float wa    = sqrtf(area / ratio);
  float ha    = wa * ratio;

  float d0 = (float)rp[(size_t)(a * 4 + 0) * hwA + s];
  float d1 = (float)rp[(size_t)(a * 4 + 1) * hwA + s];
  float d2 = (float)rp[(size_t)(a * 4 + 2) * hwA + s];
  float d3 = (float)rp[(size_t)(a * 4 + 3) * hwA + s];

  float pcx = sx + d0 * 0.1f * wa;
  float pcy = sy + d1 * 0.1f * ha;
  float pw  = __expf(d2 * 0.2f) * wa;
  float ph  = __expf(d3 * 0.2f) * ha;

  float x1 = fmaxf(pcx - 0.5f * pw, 0.0f);
  float y1 = fmaxf(pcy - 0.5f * ph, 0.0f);
  float x2 = fminf(pcx + 0.5f * pw, 512.0f);
  float y2 = fminf(pcy + 0.5f * ph, 512.0f);

  boxes[(size_t)n * 4 + 0] = x1;
  boxes[(size_t)n * 4 + 1] = y1;
  boxes[(size_t)n * 4 + 2] = x2;
  boxes[(size_t)n * 4 + 3] = y2;

  float sc = 1.0f / (1.0f + __expf(-(float)cp[(size_t)a * hwA + s]));
  scores[n] = sc;
  masked[n] = (sc > SCORE_THR) ? sc : -1.0f;
}

// ---------------------------------------------------------------------------
// Top-K (K=1000) by iterated block-wide arg-max. Single workgroup, 1024 thr.
// ---------------------------------------------------------------------------
__global__ __launch_bounds__(1024)
void k_topk(const float* __restrict__ masked, float* __restrict__ pool,
            int* __restrict__ order, float* __restrict__ tops, int N)
{
  __shared__ float smax[1024];
  __shared__ int   sidx[1024];
  int tid = threadIdx.x;
  for (int i = tid; i < N; i += 1024) pool[i] = masked[i];
  __syncthreads();

  for (int t = 0; t < KTOP; ++t) {
    float bm = -1e30f; int bi = 0;
    for (int i = tid; i < N; i += 1024) {
      float v = pool[i];
      if (v > bm) { bm = v; bi = i; }
    }
    smax[tid] = bm; sidx[tid] = bi;
    __syncthreads();
#pragma unroll
    for (int sft = 512; sft > 0; sft >>= 1) {
      if (tid < sft) {
        float vo = smax[tid + sft]; int io = sidx[tid + sft];
        if (vo > smax[tid] || (vo == smax[tid] && io < sidx[tid])) {
          smax[tid] = vo; sidx[tid] = io;
        }
      }
      __syncthreads();
    }
    if (tid == 0) {
      order[t] = sidx[0];
      tops[t]  = smax[0];
      pool[sidx[0]] = -1e30f;
    }
    __syncthreads();
  }
}

// ---------------------------------------------------------------------------
// Greedy NMS over the fixed top-K pool. Single workgroup, 1024 threads.
// ---------------------------------------------------------------------------
__global__ __launch_bounds__(1024)
void k_nms(const float* __restrict__ boxes, const int* __restrict__ order,
           const float* __restrict__ tops, float* __restrict__ tb,
           int* __restrict__ keep)
{
  int tid = threadIdx.x;
  __shared__ float bix1, biy1, bix2, biy2, biarea;
  __shared__ int   keep_i;

  for (int i = tid; i < KTOP; i += 1024) {
    int o = order[i];
    tb[i * 4 + 0] = boxes[(size_t)o * 4 + 0];
    tb[i * 4 + 1] = boxes[(size_t)o * 4 + 1];
    tb[i * 4 + 2] = boxes[(size_t)o * 4 + 2];
    tb[i * 4 + 3] = boxes[(size_t)o * 4 + 3];
    keep[i] = (tops[i] > SCORE_THR) ? 1 : 0;
  }
  __syncthreads();

  for (int i = 0; i < KTOP; ++i) {
    if (tid == 0) {
      keep_i = keep[i];
      bix1 = tb[i * 4 + 0]; biy1 = tb[i * 4 + 1];
      bix2 = tb[i * 4 + 2]; biy2 = tb[i * 4 + 3];
      biarea = (bix2 - bix1) * (biy2 - biy1);
    }
    __syncthreads();
    if (keep_i) {
      for (int j = tid; j < KTOP; j += 1024) {
        if (j > i && keep[j]) {
          float xx1 = fmaxf(bix1, tb[j * 4 + 0]);
          float yy1 = fmaxf(biy1, tb[j * 4 + 1]);
          float xx2 = fminf(bix2, tb[j * 4 + 2]);
          float yy2 = fminf(biy2, tb[j * 4 + 3]);
          float inter = fmaxf(xx2 - xx1, 0.0f) * fmaxf(yy2 - yy1, 0.0f);
          float aj = (tb[j * 4 + 2] - tb[j * 4 + 0]) * (tb[j * 4 + 3] - tb[j * 4 + 1]);
          float iou = inter / (biarea + aj - inter + 1e-9f);
          if (iou > IOU_THR) keep[j] = 0;
        }
      }
    }
    __syncthreads();
  }
}

// ---------------------------------------------------------------------------
// Assemble d_out = [kp | nms_scores | nms_class | nms_boxes | keep]
// ---------------------------------------------------------------------------
__global__ void k_finalize(const __bf16* __restrict__ kp, const float* __restrict__ scores,
                           const int* __restrict__ order, const float* __restrict__ tb,
                           const int* __restrict__ keep, float* __restrict__ out)
{
  int i = blockIdx.x * blockDim.x + threadIdx.x;
  const int total = KP_SZ + KTOP * 7;
  if (i >= total) return;
  if (i < KP_SZ) { out[i] = (float)kp[i]; return; }
  int j = i - KP_SZ;
  if (j < KTOP) { out[i] = keep[j] ? scores[order[j]] : 0.0f; return; }
  j -= KTOP;
  if (j < KTOP) { out[i] = keep[j] ? 0.0f : -1.0f; return; }   // single class -> argmax 0
  j -= KTOP;
  if (j < 4 * KTOP) { out[i] = keep[j >> 2] ? tb[j] : 0.0f; return; }
  j -= 4 * KTOP;
  out[i] = keep[j] ? 1.0f : 0.0f;
}

// ---------------------------------------------------------------------------
// Host orchestration
// ---------------------------------------------------------------------------
static inline void conv(hipStream_t s, const __bf16* X, const __bf16* Wt, const float* B,
                        __bf16* Y, int C, int D, int OC, int k3, int relu)
{
  dim3 g((D * D + BM - 1) / BM, (OC + BN - 1) / BN);
  if (k3) {
    switch (D) {
      case 128: k_conv_wmma_t<3,3,7><<<g, CONV_THREADS, 0, s>>>(X, Wt, B, Y, C, OC, relu); break;
      case  64: k_conv_wmma_t<3,3,6><<<g, CONV_THREADS, 0, s>>>(X, Wt, B, Y, C, OC, relu); break;
      case  32: k_conv_wmma_t<3,3,5><<<g, CONV_THREADS, 0, s>>>(X, Wt, B, Y, C, OC, relu); break;
      case  16: k_conv_wmma_t<3,3,4><<<g, CONV_THREADS, 0, s>>>(X, Wt, B, Y, C, OC, relu); break;
      case   8: k_conv_wmma_t<3,3,3><<<g, CONV_THREADS, 0, s>>>(X, Wt, B, Y, C, OC, relu); break;
      default:  k_conv_wmma_t<3,3,2><<<g, CONV_THREADS, 0, s>>>(X, Wt, B, Y, C, OC, relu); break;
    }
  } else {
    k_conv_wmma_t<1,1,7><<<g, CONV_THREADS, 0, s>>>(X, Wt, B, Y, C, OC, relu);
  }
}

static inline void cvt(hipStream_t s, const float* in, __bf16* out, int n)
{
  int n4 = n / 4;
  k_cvt_bf16<<<(n4 + 255) / 256, 256, 0, s>>>(in, out, n4);
}

extern "C" void kernel_launch(void* const* d_in, const int* in_sizes, int n_in,
                              void* d_out, int out_size, void* d_ws, size_t ws_size,
                              hipStream_t stream)
{
  if (n_in < 50) return;
  // setup_inputs() dict order:
  // 0:x 1:p2 2:p3 3:p4 4:p5 5:f3 6:f4 7:f5 8:f6 9:f7
  // 10..17: convt1..4 (_w,_b)  18..25: convs1..4 (_w,_b)
  // 26:conv2_w 27:conv2_b 28:convfin_w 29:convfin_b
  // 30..34: reg_ws[0..4] 35..39: reg_bs 40..44: cls_ws 45..49: cls_bs
  const float* pin[4] = {(const float*)d_in[1], (const float*)d_in[2],
                         (const float*)d_in[3], (const float*)d_in[4]}; // p2..p5
  const float* fin[5] = {(const float*)d_in[5], (const float*)d_in[6],
                         (const float*)d_in[7], (const float*)d_in[8],
                         (const float*)d_in[9]};                        // f3..f7
  const float* convt_wf[4] = {(const float*)d_in[10], (const float*)d_in[12],
                              (const float*)d_in[14], (const float*)d_in[16]};
  const float* convt_b[4]  = {(const float*)d_in[11], (const float*)d_in[13],
                              (const float*)d_in[15], (const float*)d_in[17]};
  const float* convs_wf[4] = {(const float*)d_in[18], (const float*)d_in[20],
                              (const float*)d_in[22], (const float*)d_in[24]};
  const float* convs_b[4]  = {(const float*)d_in[19], (const float*)d_in[21],
                              (const float*)d_in[23], (const float*)d_in[25]};
  const float* conv2_wf    = (const float*)d_in[26];
  const float* conv2_b     = (const float*)d_in[27];
  const float* convfin_wf  = (const float*)d_in[28];
  const float* convfin_b   = (const float*)d_in[29];
  const float* reg_wf[5]; const float* reg_b[5];
  const float* cls_wf[5]; const float* cls_b[5];
  for (int i = 0; i < 5; ++i) {
    reg_wf[i] = (const float*)d_in[30 + i];
    reg_b[i]  = (const float*)d_in[35 + i];
    cls_wf[i] = (const float*)d_in[40 + i];
    cls_b[i]  = (const float*)d_in[45 + i];
  }

  // ---- workspace layout (byte allocator, 256B aligned regions) ----
  char* base = (char*)d_ws;
  size_t off = 0;
  auto allocB = [&](size_t bytes) {
    void* p = base + off;
    off += (bytes + 255) & ~(size_t)255;
    return p;
  };
  auto allocBF = [&](size_t n) { return (__bf16*)allocB(n * sizeof(__bf16)); };
  auto allocF  = [&](size_t n) { return (float*)allocB(n * sizeof(float)); };

  // bf16 copies of input features
  const int psz[4] = {4194304, 1048576, 262144, 65536};      // p2..p5
  const int fsz[5] = {1048576, 262144, 65536, 16384, 4096};  // f3..f7
  __bf16* pb[4]; __bf16* fb[5];
  for (int i = 0; i < 4; ++i) pb[i] = allocBF(psz[i]);
  for (int i = 0; i < 5; ++i) fb[i] = allocBF(fsz[i]);
  // bf16 copies of weights
  __bf16* convt_w[4]; __bf16* convs_w[4];
  for (int i = 0; i < 4; ++i) convt_w[i] = allocBF(294912);
  for (int i = 0; i < 4; ++i) convs_w[i] = allocBF(147456);
  __bf16* conv2_w   = allocBF(1179648);
  __bf16* convfin_w = allocBF(4608);
  __bf16* reg_w[5]; __bf16* cls_w[5];
  const int rwsz[5] = {589824, 589824, 589824, 589824, 82944};
  const int cwsz[5] = {589824, 589824, 589824, 589824, 20736};
  for (int i = 0; i < 5; ++i) reg_w[i] = allocBF(rwsz[i]);
  for (int i = 0; i < 5; ++i) cls_w[i] = allocBF(cwsz[i]);
  // bf16 activations
  __bf16* ttmp = allocBF(2097152);
  __bf16* k5   = allocBF(32768);
  __bf16* k4   = allocBF(131072);
  __bf16* k3   = allocBF(524288);
  __bf16* k2   = allocBF(2097152);
  __bf16* cat  = allocBF(8388608);
  __bf16* c2   = allocBF(4194304);
  __bf16* kpb  = allocBF(294912);
  __bf16* hA   = allocBF(1048576);
  __bf16* hB   = allocBF(1048576);
  const int hw[5]  = {64, 32, 16, 8, 4};
  const int rsz[5] = {147456, 36864, 9216, 2304, 576};
  const int csz[5] = {36864, 9216, 2304, 576, 144};
  __bf16* regout[5]; __bf16* clsout[5];
  for (int i = 0; i < 5; ++i) regout[i] = allocBF(rsz[i]);
  for (int i = 0; i < 5; ++i) clsout[i] = allocBF(csz[i]);
  // f32 tail buffers
  float* boxes  = allocF((size_t)NANCH * 4);
  float* scores = allocF(NANCH);
  float* masked = allocF(NANCH);
  float* pool   = allocF(NANCH);
  float* tops   = allocF(KTOP);
  float* tb     = allocF(KTOP * 4);
  int*   order  = (int*)allocB(KTOP * sizeof(int));
  int*   keep   = (int*)allocB(KTOP * sizeof(int));
  (void)ws_size; (void)in_sizes; (void)out_size;

  // ---- convert inputs + weights to bf16 (one pass each) ----
  for (int i = 0; i < 4; ++i) cvt(stream, pin[i], pb[i], psz[i]);
  for (int i = 0; i < 5; ++i) cvt(stream, fin[i], fb[i], fsz[i]);
  for (int i = 0; i < 4; ++i) cvt(stream, convt_wf[i], convt_w[i], 294912);
  for (int i = 0; i < 4; ++i) cvt(stream, convs_wf[i], convs_w[i], 147456);
  cvt(stream, conv2_wf, conv2_w, 1179648);
  cvt(stream, convfin_wf, convfin_w, 4608);
  for (int i = 0; i < 5; ++i) cvt(stream, reg_wf[i], reg_w[i], rwsz[i]);
  for (int i = 0; i < 5; ++i) cvt(stream, cls_wf[i], cls_w[i], cwsz[i]);

  // ---- keypoint branch: convt -> convs (no relu), per level ----
  const __bf16* kin[4]  = {pb[3], pb[2], pb[1], pb[0]};  // p5, p4, p3, p2
  __bf16*       kout[4] = {k5, k4, k3, k2};
  const int     kdim[4] = {16, 32, 64, 128};
  for (int i = 0; i < 4; ++i) {
    int d = kdim[i];
    conv(stream, kin[i], convt_w[i], convt_b[i], ttmp, 256, d, 128, 1, 0);
    conv(stream, ttmp,   convs_w[i], convs_b[i], kout[i], 128, d, 128, 1, 0);
  }
  {
    int total = 512 * 128 * 128;
    k_upcat<<<(total + 255) / 256, 256, 0, stream>>>(k5, k4, k3, k2, cat);
  }
  conv(stream, cat, conv2_w, conv2_b, c2, 512, 128, 256, 1, 1);      // 3x3 + relu
  conv(stream, c2, convfin_w, convfin_b, kpb, 256, 128, 18, 0, 0);   // 1x1, pad 0

  // ---- detection heads (reg: OC 36, cls: OC 9 raw logits) ----
  for (int lv = 0; lv < 5; ++lv) {
    int d = hw[lv];
    // reg subnet
    conv(stream, fb[lv], reg_w[0], reg_b[0], hA, 256, d, 256, 1, 1);
    conv(stream, hA,     reg_w[1], reg_b[1], hB, 256, d, 256, 1, 1);
    conv(stream, hB,     reg_w[2], reg_b[2], hA, 256, d, 256, 1, 1);
    conv(stream, hA,     reg_w[3], reg_b[3], hB, 256, d, 256, 1, 1);
    conv(stream, hB,     reg_w[4], reg_b[4], regout[lv], 256, d, 36, 1, 0);
    // cls subnet
    conv(stream, fb[lv], cls_w[0], cls_b[0], hA, 256, d, 256, 1, 1);
    conv(stream, hA,     cls_w[1], cls_b[1], hB, 256, d, 256, 1, 1);
    conv(stream, hB,     cls_w[2], cls_b[2], hA, 256, d, 256, 1, 1);
    conv(stream, hA,     cls_w[3], cls_b[3], hB, 256, d, 256, 1, 1);
    conv(stream, hB,     cls_w[4], cls_b[4], clsout[lv], 256, d, 9, 1, 0);
  }

  // ---- decode + topK + NMS + output ----
  k_decode<<<(NANCH + 255) / 256, 256, 0, stream>>>(
      regout[0], regout[1], regout[2], regout[3], regout[4],
      clsout[0], clsout[1], clsout[2], clsout[3], clsout[4],
      boxes, scores, masked);
  k_topk<<<1, 1024, 0, stream>>>(masked, pool, order, tops, NANCH);
  k_nms<<<1, 1024, 0, stream>>>(boxes, order, tops, tb, keep);
  {
    int total = KP_SZ + KTOP * 7;
    k_finalize<<<(total + 255) / 256, 256, 0, stream>>>(kpb, scores, order, tb, keep,
                                                        (float*)d_out);
  }
}
